// GATBackbone_35493609734453
// MI455X (gfx1250) — compile-verified, matrix operand order checked by reading, not run on hardware
//
#include <hip/hip_runtime.h>
#include <cmath>

#define Nn 40000
#define Ee 640000
#define ET (Ee + Nn)          // 680000 edges incl. self-loops
#define Dd 128
#define Hh 2
#define Cc 128
#define HC (Hh * Cc)          // 256
#define Ll 2
#define SLOPE 0.2f

typedef float v2f  __attribute__((ext_vector_type(2)));
typedef float v8f  __attribute__((ext_vector_type(8)));
typedef __bf16 v16bf __attribute__((ext_vector_type(16)));

#if __has_builtin(__builtin_amdgcn_wmma_f32_16x16x4_f32)
#define HAVE_WMMA_F32 1
#else
#define HAVE_WMMA_F32 0
#endif

// ---- helpers -------------------------------------------------------------
__device__ __forceinline__ unsigned fkey(float f) {           // order-preserving float->uint
  unsigned u = __float_as_uint(f);
  return (u & 0x80000000u) ? ~u : (u | 0x80000000u);
}
__device__ __forceinline__ float funkey(unsigned k) {
  return __uint_as_float((k & 0x80000000u) ? (k ^ 0x80000000u) : ~k);
}
__device__ __forceinline__ void atomAddF(float* p, float v) { // non-returning native f32 add
  __hip_atomic_fetch_add(p, v, __ATOMIC_RELAXED, __HIP_MEMORY_SCOPE_AGENT);
}
__device__ __forceinline__ void atomMaxU(unsigned* p, unsigned v) {
  __hip_atomic_fetch_max(p, v, __ATOMIC_RELAXED, __HIP_MEMORY_SCOPE_AGENT);
}
__device__ __forceinline__ float wred(float v) {
  #pragma unroll
  for (int o = 16; o > 0; o >>= 1) v += __shfl_xor(v, o, 32);
  return v;
}
__device__ __forceinline__ float dot4(float4 a, float4 b) {
  return a.x * b.x + a.y * b.y + a.z * b.z + a.w * b.w;
}

// ---- init: zero accumulator / softmax state ------------------------------
__global__ __launch_bounds__(256) void k_init(float* __restrict__ accum,
                                              unsigned* __restrict__ mk,
                                              float* __restrict__ ssum) {
  int i = blockIdx.x * 256 + threadIdx.x;
  if (i < Nn * HC) accum[i] = 0.0f;
  if (i < Nn * Hh) { mk[i] = 0u; ssum[i] = 0.0f; }   // key(x) > 0 for all finite x
}

// ---- h = x @ W^T via WMMA; one wave per 16x16 output tile ----------------
__global__ __launch_bounds__(256) void k_gemm(const float* __restrict__ X,
                                              const float* __restrict__ Wl,
                                              float* __restrict__ Hf) {
  const int lane = threadIdx.x & 31;
  const int tile = blockIdx.x * 8 + (threadIdx.x >> 5); // 2500*16 tiles, exact
  const int mt = tile >> 4;
  const int ot = tile & 15;
  const int g  = lane >> 4;          // half-wave select
  const int ln = lane & 15;
  const int row = mt * 16 + ln;      // A: M = lane&15
  const int oc  = ot * 16 + ln;      // B: N = lane&15
  v8f acc = {};
#if HAVE_WMMA_F32
  const int kb = g * 2;              // lanes 16-31 hold K+2,K+3
  for (int k0 = 0; k0 < Dd; k0 += 4) {
    v2f a = *(const v2f*)(X  + row * Dd + k0 + kb);
    v2f b = *(const v2f*)(Wl + oc  * Dd + k0 + kb);
    acc = __builtin_amdgcn_wmma_f32_16x16x4_f32(false, a, false, b,
                                                (short)0, acc, false, false);
  }
#else
  for (int k0 = 0; k0 < Dd; k0 += 32) {
    v16bf a, b;
    #pragma unroll
    for (int e = 0; e < 16; ++e) {
      // 16-bit A layout: lanes0-15 K=0..7 (v0..3), K=16..23 (v4..7); lanes16-31 +8
      int k = k0 + 8 * g + e + ((e < 8) ? 0 : 8);
      a[e] = (__bf16)X[row * Dd + k];
      b[e] = (__bf16)Wl[oc * Dd + k];
    }
    acc = __builtin_amdgcn_wmma_f32_16x16x32_bf16(false, a, false, b,
                                                  (short)0, acc, false, false);
  }
#endif
  #pragma unroll
  for (int r = 0; r < 8; ++r)        // C/D: VGPR r -> M = r + 8*g, N = lane&15
    Hf[(mt * 16 + g * 8 + r) * HC + oc] = acc[r];
}

// ---- per-node attention logits: one wave per node ------------------------
__global__ __launch_bounds__(256) void k_logits(const float* __restrict__ Hf,
                                                const float* __restrict__ AS,
                                                const float* __restrict__ AD,
                                                float* __restrict__ esrc,
                                                float* __restrict__ edst) {
  const int lane = threadIdx.x & 31;
  const int n = (blockIdx.x * blockDim.x + threadIdx.x) >> 5;
  if (n >= Nn) return;
  const float4 h0  = *(const float4*)(Hf + (size_t)n * HC + lane * 4);
  const float4 h1  = *(const float4*)(Hf + (size_t)n * HC + Cc + lane * 4);
  const float4 as0 = *(const float4*)(AS + lane * 4);
  const float4 as1 = *(const float4*)(AS + Cc + lane * 4);
  const float4 ad0 = *(const float4*)(AD + lane * 4);
  const float4 ad1 = *(const float4*)(AD + Cc + lane * 4);
  float s0 = wred(dot4(h0, as0));
  float s1 = wred(dot4(h1, as1));
  float d0 = wred(dot4(h0, ad0));
  float d1 = wred(dot4(h1, ad1));
  if (lane == 0) {
    esrc[n * Hh + 0] = s0; esrc[n * Hh + 1] = s1;
    edst[n * Hh + 0] = d0; edst[n * Hh + 1] = d1;
  }
}

// ---- segment max of leaky-relu logits over dst ---------------------------
__global__ __launch_bounds__(256) void k_edge_max(const int* __restrict__ EI,
                                                  const float* __restrict__ esrc,
                                                  const float* __restrict__ edst,
                                                  unsigned* __restrict__ mk) {
  int e = blockIdx.x * blockDim.x + threadIdx.x;
  if (e >= ET) return;
  int s = (e < Ee) ? EI[e]      : (e - Ee);
  int d = (e < Ee) ? EI[Ee + e] : (e - Ee);
  #pragma unroll
  for (int h = 0; h < Hh; ++h) {
    float v = esrc[s * Hh + h] + edst[d * Hh + h];
    v = (v >= 0.0f) ? v : SLOPE * v;
    atomMaxU(&mk[d * Hh + h], fkey(v));
  }
}

// ---- p = exp(e - m[dst]); s[dst] += p ------------------------------------
__global__ __launch_bounds__(256) void k_edge_p(const int* __restrict__ EI,
                                                const float* __restrict__ esrc,
                                                const float* __restrict__ edst,
                                                const unsigned* __restrict__ mk,
                                                float* __restrict__ pbuf,
                                                float* __restrict__ ssum) {
  int e = blockIdx.x * blockDim.x + threadIdx.x;
  if (e >= ET) return;
  int s = (e < Ee) ? EI[e]      : (e - Ee);
  int d = (e < Ee) ? EI[Ee + e] : (e - Ee);
  #pragma unroll
  for (int h = 0; h < Hh; ++h) {
    float v = esrc[s * Hh + h] + edst[d * Hh + h];
    v = (v >= 0.0f) ? v : SLOPE * v;
    float pv = expf(v - funkey(mk[d * Hh + h]));
    pbuf[e * Hh + h] = pv;
    atomAddF(&ssum[d * Hh + h], pv);
  }
}

// ---- weighted scatter-add: one wave per (edge, head), 4 ch per lane ------
__global__ __launch_bounds__(256) void k_agg(const int* __restrict__ EI,
                                             const float* __restrict__ Hf,
                                             const float* __restrict__ pbuf,
                                             const float* __restrict__ ssum,
                                             float* __restrict__ accum) {
  const int lane = threadIdx.x & 31;
  const int w = (blockIdx.x * blockDim.x + threadIdx.x) >> 5;
  if (w >= ET * Hh) return;
  const int e = w >> 1;
  const int h = w & 1;
  const int s = (e < Ee) ? EI[e]      : (e - Ee);
  const int d = (e < Ee) ? EI[Ee + e] : (e - Ee);
  const float coef = pbuf[e * Hh + h] / ssum[d * Hh + h];
  const float4 hv = *(const float4*)(Hf + (size_t)s * HC + h * Cc + lane * 4);
  float* op = accum + (size_t)d * HC + h * Cc + lane * 4;
  atomAddF(op + 0, coef * hv.x);
  atomAddF(op + 1, coef * hv.y);
  atomAddF(op + 2, coef * hv.z);
  atomAddF(op + 3, coef * hv.w);
}

// ---- mean over heads + bias + ELU ----------------------------------------
__global__ __launch_bounds__(256) void k_final(const float* __restrict__ accum,
                                               const float* __restrict__ bias,
                                               float* __restrict__ xout) {
  int i = blockIdx.x * blockDim.x + threadIdx.x;
  if (i >= Nn * Dd) return;
  int n = i >> 7, c = i & 127;
  float v = 0.5f * (accum[(size_t)n * HC + c] + accum[(size_t)n * HC + Cc + c]) + bias[c];
  xout[i] = (v > 0.0f) ? v : expm1f(v);
}

// ---- driver --------------------------------------------------------------
extern "C" void kernel_launch(void* const* d_in, const int* in_sizes, int n_in,
                              void* d_out, int out_size, void* d_ws, size_t ws_size,
                              hipStream_t stream) {
  (void)in_sizes; (void)n_in; (void)out_size; (void)ws_size;
  const float* x  = (const float*)d_in[0];
  const int*   EI = (const int*)d_in[1];     // [2, E]
  const float* W  = (const float*)d_in[2];   // [L, 256, 128]
  const float* AS = (const float*)d_in[3];   // [L, 2, 128]
  const float* AD = (const float*)d_in[4];   // [L, 2, 128]
  const float* B  = (const float*)d_in[5];   // [L, 128]

  float*    Hf    = (float*)d_ws;                          // N*256
  float*    accum = Hf + (size_t)Nn * HC;                  // N*256
  float*    esrc  = accum + (size_t)Nn * HC;               // N*2
  float*    edst  = esrc + Nn * Hh;                        // N*2
  unsigned* mk    = (unsigned*)(edst + Nn * Hh);           // N*2
  float*    ssum  = (float*)(mk + Nn * Hh);                // N*2
  float*    pbuf  = ssum + Nn * Hh;                        // ET*2

  float* out0 = (float*)d_out;
  hipMemcpyAsync(out0, x, (size_t)Nn * Dd * sizeof(float),
                 hipMemcpyDeviceToDevice, stream);

  const float* xin = x;
  for (int l = 0; l < Ll; ++l) {
    float* xout = out0 + (size_t)(l + 1) * Nn * Dd;
    k_init    <<<(Nn * HC + 255) / 256, 256, 0, stream>>>(accum, mk, ssum);
    k_gemm    <<<(Nn / 16) * 16 / 8,    256, 0, stream>>>(xin, W + (size_t)l * HC * Dd, Hf);
    k_logits  <<<(Nn * 32 + 255) / 256, 256, 0, stream>>>(Hf, AS + l * Hh * Cc,
                                                          AD + l * Hh * Cc, esrc, edst);
    k_edge_max<<<(ET + 255) / 256,      256, 0, stream>>>(EI, esrc, edst, mk);
    k_edge_p  <<<(ET + 255) / 256,      256, 0, stream>>>(EI, esrc, edst, mk, pbuf, ssum);
    k_agg     <<<(ET * Hh * 32 + 255) / 256, 256, 0, stream>>>(EI, Hf, pbuf, ssum, accum);
    k_final   <<<(Nn * Dd + 255) / 256, 256, 0, stream>>>(accum, B + l * Dd, xout);
    xin = xout;
  }
}
